// WeightOnlyInt8Linear_43928925503621
// MI455X (gfx1250) — matrix operand order repaired and checked by reading, not compiled, not run on hardware
//
#include <hip/hip_runtime.h>
#include <hip/hip_bf16.h>
#include <stdint.h>

// Shapes (fixed by the reference)
#define M_TOTAL 8192   // B*S = 4*2048
#define K_TOTAL 4096   // IN
#define N_TOTAL 4096   // OUT

#define LDS_STRIDE 40  // 32 + 8 pad elements -> 80B row stride (16B aligned, spreads 64 banks)

typedef __attribute__((ext_vector_type(16))) __bf16 v16bf;
typedef __attribute__((ext_vector_type(8)))  __bf16 v8bf;
typedef __attribute__((ext_vector_type(8)))  float  v8f;

union V16 { v16bf v; v8bf h[2]; };

// NOTE: the gfx1250 ASYNCcnt global->LDS path (__builtin_amdgcn_global_load_async_to_lds_b128)
// compiles here, but it lets the scheduler pipeline fragment ds_loads several iterations deep:
// VGPR usage blew past 256 (s_set_vgpr_msb mode) with heavy v_mov_b64 copy churn, destroying
// occupancy. The synchronous register-staged copy below produced strictly better codegen
// (~214 VGPRs, clause'd loads, no copies), so async staging is intentionally disabled.

// ---------------------------------------------------------------------------
// Kernel 1: one-shot weight dequant  int8 -> bf16 (exact; |w|<=127 fits bf16)
// 16 MB reads + 32 MB writes, pure streaming; weight then lives in L2 (192MB).
// ---------------------------------------------------------------------------
__device__ __forceinline__ void cvt16_i8_bf16(uint4 raw, v8bf& lo, v8bf& hi) {
    uint32_t w[4] = {raw.x, raw.y, raw.z, raw.w};
#pragma unroll
    for (int i = 0; i < 4; ++i) {
#pragma unroll
        for (int j = 0; j < 4; ++j) {
            const int idx = i * 4 + j;
            __bf16 v = (__bf16)(float)(int)(int8_t)((w[i] >> (8 * j)) & 0xff);
            if (idx < 8) lo[idx] = v; else hi[idx - 8] = v;
        }
    }
}

__global__ __launch_bounds__(256)
void dequant_w_i8_bf16(const int8_t* __restrict__ W, __bf16* __restrict__ Wb) {
    const size_t i = ((size_t)blockIdx.x * 256 + threadIdx.x) * 16;
    uint4 raw = *(const uint4*)(W + i);
    v8bf lo, hi;
    cvt16_i8_bf16(raw, lo, hi);
    *(v8bf*)(Wb + i)     = lo;
    *(v8bf*)(Wb + i + 8) = hi;
}

// ---------------------------------------------------------------------------
// Kernel 2: pure bf16 GEMM, block 256(M) x 128(N) x 32(K), 8 waves,
// wave tile 64x64 -> 16 WMMAs per K-step per wave (1:1 with ds_load_b128).
// ---------------------------------------------------------------------------
#define GBM 256
#define GBN 128
#define GBK 32
#define GKT (K_TOTAL / GBK)

__global__ __launch_bounds__(256)
void gemm_bf16_wmma(const uint16_t* __restrict__ X,          // bf16 bits [M,K]
                    const __bf16*   __restrict__ Wb,         // bf16 [N,K] (dequantized)
                    const __hip_bfloat16* __restrict__ SCL,  // bf16 [N]
                    __hip_bfloat16* __restrict__ Y)          // bf16 [M,N]
{
    __shared__ __bf16 lA[2][GBM * LDS_STRIDE];   // 2 x 20 KB
    __shared__ __bf16 lB[2][GBN * LDS_STRIDE];   // 2 x 10 KB

    const int tid  = threadIdx.x;
    const int lane = tid & 31;
    const int wv   = tid >> 5;      // 0..7
    const int wm   = wv >> 1;       // 0..3 -> 64-row slab
    const int wn   = wv & 1;        // 0..1 -> 64-col slab
    const int l15  = lane & 15;
    const int lhi  = lane >> 4;     // 0 or 1

    const int m0 = blockIdx.x * GBM;
    const int n0 = blockIdx.y * GBN;

    // Staging: thread owns rows (srow, srow+128) of A (32B chunks) and row srow of B
    const int srow  = tid >> 1;            // 0..127
    const int skoff = (tid & 1) * 16;      // element offset within GBK
    const int sidxA0 = srow * LDS_STRIDE + skoff;
    const int sidxA1 = (srow + 128) * LDS_STRIDE + skoff;
    const int sidxB  = srow * LDS_STRIDE + skoff;

    const uint16_t* xg0 = X  + (size_t)(m0 + srow)       * K_TOTAL + skoff;
    const uint16_t* xg1 = X  + (size_t)(m0 + srow + 128) * K_TOTAL + skoff;
    const uint16_t* wg  = (const uint16_t*)Wb + (size_t)(n0 + srow) * K_TOTAL + skoff;

    v8f acc[4][4] = {};

    // ---- prologue: stage 0 ----
    {
        uint4 a0 = *(const uint4*)(xg0);
        uint4 a1 = *(const uint4*)(xg0 + 8);
        uint4 a2 = *(const uint4*)(xg1);
        uint4 a3 = *(const uint4*)(xg1 + 8);
        uint4 b0 = *(const uint4*)(wg);
        uint4 b1 = *(const uint4*)(wg + 8);
        *(uint4*)&lA[0][sidxA0]     = a0;  *(uint4*)&lA[0][sidxA0 + 8] = a1;
        *(uint4*)&lA[0][sidxA1]     = a2;  *(uint4*)&lA[0][sidxA1 + 8] = a3;
        *(uint4*)&lB[0][sidxB]      = b0;  *(uint4*)&lB[0][sidxB + 8]  = b1;
    }
    __syncthreads();

    for (int kt = 0; kt < GKT; ++kt) {
        const int cur = kt & 1;
        const int nxt = cur ^ 1;
        const bool pf = (kt + 1) < GKT;

        uint4 a0, a1, a2, a3, b0, b1;
        if (pf) {
            const uint16_t* xp0 = xg0 + (size_t)(kt + 1) * GBK;
            const uint16_t* xp1 = xg1 + (size_t)(kt + 1) * GBK;
            const uint16_t* wp  = wg  + (size_t)(kt + 1) * GBK;
            a0 = *(const uint4*)(xp0);  a1 = *(const uint4*)(xp0 + 8);
            a2 = *(const uint4*)(xp1);  a3 = *(const uint4*)(xp1 + 8);
            b0 = *(const uint4*)(wp);   b1 = *(const uint4*)(wp + 8);
        }

        // A fragments (16-bit 16x32 A layout: lane<16 -> K[0..7]&K[16..23])
        V16 af[4];
#pragma unroll
        for (int mt = 0; mt < 4; ++mt) {
            const __bf16* p = &lA[cur][(wm * 64 + mt * 16 + l15) * LDS_STRIDE + lhi * 8];
            af[mt].h[0] = *(const v8bf*)(p);
            af[mt].h[1] = *(const v8bf*)(p + 16);
        }
        // B fragments (32x16 B layout: lane<16 -> K[0..7]&K[16..23])
        V16 bb[4];
#pragma unroll
        for (int nt = 0; nt < 4; ++nt) {
            const __bf16* p = &lB[cur][(wn * 64 + nt * 16 + l15) * LDS_STRIDE + lhi * 8];
            bb[nt].h[0] = *(const v8bf*)(p);
            bb[nt].h[1] = *(const v8bf*)(p + 16);
        }

        // Write next stage (overlaps with WMMA issue)
        if (pf) {
            *(uint4*)&lA[nxt][sidxA0]     = a0;  *(uint4*)&lA[nxt][sidxA0 + 8] = a1;
            *(uint4*)&lA[nxt][sidxA1]     = a2;  *(uint4*)&lA[nxt][sidxA1 + 8] = a3;
            *(uint4*)&lB[nxt][sidxB]      = b0;  *(uint4*)&lB[nxt][sidxB + 8]  = b1;
        }

        // 16 WMMAs per wave per K-step
#pragma unroll
        for (int mt = 0; mt < 4; ++mt)
#pragma unroll
            for (int nt = 0; nt < 4; ++nt)
                acc[mt][nt] = __builtin_amdgcn_wmma_f32_16x16x32_bf16(
                    false, af[mt].v, false, bb[nt].v,
                    (short)0, acc[mt][nt], false, false);

        __syncthreads();
    }

    // ---- epilogue: per-channel scale + f32 -> bf16 ----
#pragma unroll
    for (int nt = 0; nt < 4; ++nt) {
        const int col = n0 + wn * 64 + nt * 16 + l15;
        const float sf = __bfloat162float(SCL[col]);
#pragma unroll
        for (int mt = 0; mt < 4; ++mt) {
#pragma unroll
            for (int r = 0; r < 8; ++r) {
                const int row = m0 + wm * 64 + mt * 16 + lhi * 8 + r;
                Y[(size_t)row * N_TOTAL + col] = __float2bfloat16(acc[mt][nt][r] * sf);
            }
        }
    }
}

// ---------------------------------------------------------------------------
// Fallback: fused int8->bf16 GEMM (round-1 kernel), used if ws too small.
// ---------------------------------------------------------------------------
#define FBM 128
#define FBN 128
#define FBK 32
#define FKT (K_TOTAL / FBK)

__global__ __launch_bounds__(256)
void gemm_fused_i8_wmma(const uint16_t* __restrict__ X,
                        const int8_t*   __restrict__ W,
                        const __hip_bfloat16* __restrict__ SCL,
                        __hip_bfloat16* __restrict__ Y)
{
    __shared__ __bf16 lA[2][FBM * LDS_STRIDE];
    __shared__ __bf16 lB[2][FBN * LDS_STRIDE];

    const int tid  = threadIdx.x;
    const int lane = tid & 31;
    const int wv   = tid >> 5;
    const int wm   = wv >> 1;       // 0..3 -> 32-row slab
    const int wn   = wv & 1;        // 0..1 -> 64-col slab
    const int l15  = lane & 15;
    const int lhi  = lane >> 4;

    const int m0 = blockIdx.x * FBM;
    const int n0 = blockIdx.y * FBN;

    const int srow  = tid >> 1;
    const int skoff = (tid & 1) * 16;
    const int sidx  = srow * LDS_STRIDE + skoff;

    const uint16_t* xg = X + (size_t)(m0 + srow) * K_TOTAL + skoff;
    const int8_t*   wg = W + (size_t)(n0 + srow) * K_TOTAL + skoff;

    v8f acc[2][4] = {};

    {
        uint4 a0 = *(const uint4*)(xg);
        uint4 a1 = *(const uint4*)(xg + 8);
        uint4 wr = *(const uint4*)(wg);
        *(uint4*)&lA[0][sidx]     = a0;
        *(uint4*)&lA[0][sidx + 8] = a1;
        v8bf lo, hi; cvt16_i8_bf16(wr, lo, hi);
        *(v8bf*)&lB[0][sidx]      = lo;
        *(v8bf*)&lB[0][sidx + 8]  = hi;
    }
    __syncthreads();

    for (int kt = 0; kt < FKT; ++kt) {
        const int cur = kt & 1;
        const int nxt = cur ^ 1;
        const bool pf = (kt + 1) < FKT;

        uint4 a0, a1, wr;
        if (pf) {
            const uint16_t* xp = xg + (size_t)(kt + 1) * FBK;
            const int8_t*   wp = wg + (size_t)(kt + 1) * FBK;
            a0 = *(const uint4*)(xp);
            a1 = *(const uint4*)(xp + 8);
            wr = *(const uint4*)(wp);
        }

        V16 af[2];
#pragma unroll
        for (int mt = 0; mt < 2; ++mt) {
            const __bf16* p = &lA[cur][(wm * 32 + mt * 16 + l15) * LDS_STRIDE + lhi * 8];
            af[mt].h[0] = *(const v8bf*)(p);
            af[mt].h[1] = *(const v8bf*)(p + 16);
        }
        V16 bb[4];
#pragma unroll
        for (int nt = 0; nt < 4; ++nt) {
            const __bf16* p = &lB[cur][(wn * 64 + nt * 16 + l15) * LDS_STRIDE + lhi * 8];
            bb[nt].h[0] = *(const v8bf*)(p);
            bb[nt].h[1] = *(const v8bf*)(p + 16);
        }

        if (pf) {
            *(uint4*)&lA[nxt][sidx]     = a0;
            *(uint4*)&lA[nxt][sidx + 8] = a1;
            v8bf lo, hi; cvt16_i8_bf16(wr, lo, hi);
            *(v8bf*)&lB[nxt][sidx]      = lo;
            *(v8bf*)&lB[nxt][sidx + 8]  = hi;
        }

#pragma unroll
        for (int mt = 0; mt < 2; ++mt)
#pragma unroll
            for (int nt = 0; nt < 4; ++nt)
                acc[mt][nt] = __builtin_amdgcn_wmma_f32_16x16x32_bf16(
                    false, af[mt].v, false, bb[nt].v,
                    (short)0, acc[mt][nt], false, false);

        __syncthreads();
    }

#pragma unroll
    for (int nt = 0; nt < 4; ++nt) {
        const int col = n0 + wn * 64 + nt * 16 + l15;
        const float sf = __bfloat162float(SCL[col]);
#pragma unroll
        for (int mt = 0; mt < 2; ++mt) {
#pragma unroll
            for (int r = 0; r < 8; ++r) {
                const int row = m0 + wm * 32 + mt * 16 + lhi * 8 + r;
                Y[(size_t)row * N_TOTAL + col] = __float2bfloat16(acc[mt][nt][r] * sf);
            }
        }
    }
}

// ---------------------------------------------------------------------------
extern "C" void kernel_launch(void* const* d_in, const int* in_sizes, int n_in,
                              void* d_out, int out_size, void* d_ws, size_t ws_size,
                              hipStream_t stream) {
    (void)in_sizes; (void)n_in; (void)out_size;
    const uint16_t*       x   = (const uint16_t*)d_in[0];
    const int8_t*         w   = (const int8_t*)d_in[1];
    const __hip_bfloat16* scl = (const __hip_bfloat16*)d_in[2];
    __hip_bfloat16*       y   = (__hip_bfloat16*)d_out;

    const size_t need = (size_t)N_TOTAL * K_TOTAL * sizeof(uint16_t);  // 32 MB
    if (ws_size >= need && d_ws != nullptr) {
        __bf16* wb = (__bf16*)d_ws;
        const int total16 = (N_TOTAL * K_TOTAL) / 16;                 // int8 chunks of 16
        dequant_w_i8_bf16<<<dim3(total16 / 256), dim3(256), 0, stream>>>(w, wb);
        dim3 grid(M_TOTAL / GBM, N_TOTAL / GBN, 1);                   // 32 x 32
        gemm_bf16_wmma<<<grid, dim3(256), 0, stream>>>(x, wb, scl, y);
    } else {
        dim3 grid(M_TOTAL / FBM, N_TOTAL / FBN, 1);                   // 64 x 32
        gemm_fused_i8_wmma<<<grid, dim3(256), 0, stream>>>(x, w, scl, y);
    }
}
